// MHMoE_68083821576853
// MI455X (gfx1250) — matrix operand order repaired and checked by reading, not compiled, not run on hardware
//
#include <hip/hip_runtime.h>

typedef __attribute__((ext_vector_type(16))) _Float16 v16h;
typedef __attribute__((ext_vector_type(8)))  _Float16 v8h;
typedef __attribute__((ext_vector_type(4)))  _Float16 v4h;
typedef __attribute__((ext_vector_type(8)))  float    v8f;
typedef __attribute__((ext_vector_type(4))) unsigned int u32x4;
typedef __attribute__((ext_vector_type(8))) int          i32x8;
typedef __attribute__((ext_vector_type(4))) int          i32x4;

#define LDX  264   // padded f16 row stride (528B = 512B data + 16B pad; 16B-aligned, 4-bank rotate)
#define LDH2 136
#define LDG  72

// LDS layout (bytes)
#define OFF_XH  0
#define OFF_H1  67584
#define OFF_W0  135168
#define WSTRIDE 67584           // OFF_W1 = OFF_W0 + WSTRIDE
#define OFF_H2  270336          // g1 aliases this region (gate phase only)
#define OFF_F   305152          // float scratch
#define LDS_TOTAL 315904        // < 320 KB

#if defined(__has_builtin)
#if __has_builtin(__builtin_amdgcn_tensor_load_to_lds)
#define HAVE_TDM 1
#endif
#endif

// ---------------- fragment loaders (ISA 7.12.2 layouts, wave32) ----------------
__device__ __forceinline__ v16h fragA_ld(const _Float16* base, int ld, int row0, int k0,
                                         int r, int hf) {
  const _Float16* p = base + (row0 + r) * ld + k0 + hf * 8;
  v8h lo = *(const v8h*)p;
  v8h hi = *(const v8h*)(p + 16);
  return __builtin_shufflevector(lo, hi, 0,1,2,3,4,5,6,7,8,9,10,11,12,13,14,15);
}
__device__ __forceinline__ v16h fragB_ld(const _Float16* base, int ld, int row0, int k0,
                                         int r, int hf) {
  const _Float16* p = base + (row0 + r) * ld + k0 + hf * 8;
  v8h lo = *(const v8h*)p;
  v8h hi = *(const v8h*)(p + 16);
  return __builtin_shufflevector(lo, hi, 0,1,2,3,4,5,6,7,8,9,10,11,12,13,14,15);
}
__device__ __forceinline__ v8f wmma16(v16h a, v16h b, v8f c) {
  return __builtin_amdgcn_wmma_f32_16x16x32_f16(false, a, false, b, (short)0, c, false, false);
}

// ---------------- async weight tile load: global [rows][256] f16 -> padded LDS ----------------
// TDM D# per ISA 8.3-8.6: 2D tile, data_size=2B, tile_dim0=256, tile_dim1=rows,
// dim0_stride=256, pad 4 DWORDs (16B) after every 128 DWORDs (512B row) -> LDX stride.
__device__ __forceinline__ void wgt_load_async(_Float16* dst, unsigned lds_off,
                                               const _Float16* __restrict__ src,
                                               int rows, int lane) {
#if HAVE_TDM
  (void)dst; (void)lane;
  unsigned long long ga = (unsigned long long)(uintptr_t)src;
  u32x4 g0;
  g0[0] = 1u;                                            // count=1, user mode
  g0[1] = lds_off;                                       // lds_addr (bytes)
  g0[2] = (unsigned)(ga & 0xffffffffu);                  // global_addr[31:0]
  g0[3] = (unsigned)((ga >> 32) & 0x01ffffffu) | (2u << 30);  // addr[56:32], type=2
  i32x8 g1;
  g1[0] = (int)((1u << 16) | (1u << 20) | (6u << 22) | (3u << 25)); // 2B, pad_en, int=128DW, amt=4DW
  g1[1] = (int)(256u << 16);                             // tensor_dim0 = 256
  g1[2] = (int)((unsigned)rows << 16);                   // tensor_dim1 = rows
  g1[3] = (int)(256u << 16);                             // tile_dim0 = 256
  g1[4] = rows;                                          // tile_dim1 = rows (tile_dim2=0)
  g1[5] = 256;                                           // tensor_dim0_stride = 256
  g1[6] = 0; g1[7] = 0;
  i32x4 gz = {0, 0, 0, 0};
#if __clang_major__ >= 23
  i32x8 gz8 = {0, 0, 0, 0, 0, 0, 0, 0};
  __builtin_amdgcn_tensor_load_to_lds(g0, g1, gz, gz, gz8, 0);
#else
  __builtin_amdgcn_tensor_load_to_lds(g0, g1, gz, gz, 0);
#endif
#else
  (void)lds_off;
  for (int c = lane; c < rows * 32; c += 32) {           // fallback: wave-0 copy
    int row = c >> 5, col = (c & 31) << 3;
    *(v8h*)(dst + row * LDX + col) = *(const v8h*)(src + row * 256 + col);
  }
#endif
}
__device__ __forceinline__ void wgt_wait() {
#if HAVE_TDM
  __builtin_amdgcn_s_wait_tensorcnt(0);
#endif
}

// ---------------- weight preprocessing: f32 -> f16 transposed ----------------
__global__ __launch_bounds__(256) void moe_prep_kernel(
    const float* __restrict__ gW1, const float* __restrict__ We1,
    const float* __restrict__ We2,
    _Float16* __restrict__ gW1t, _Float16* __restrict__ W1t, _Float16* __restrict__ W2t) {
  int i = blockIdx.x * 256 + threadIdx.x;
  if (i < 8 * 256 * 256) {  // We1[e][d][h] -> W1t[e][h][d]
    int e = i >> 16, rem = i & 65535, d = rem >> 8, h = rem & 255;
    W1t[(e << 16) + (h << 8) + d] = (_Float16)We1[i];
  }
  if (i < 8 * 256 * 128) {  // We2[e][h][k] -> W2t[e][k][h]
    int e = i >> 15, rem = i & 32767, h = rem >> 7, k = rem & 127;
    W2t[(e << 15) + (k << 8) + h] = (_Float16)We2[i];
  }
  if (i < 256 * 64) {       // gW1[d][j] -> gW1t[j][d]
    int d = i >> 6, j = i & 63;
    gW1t[j * 256 + d] = (_Float16)gW1[i];
  }
}

// ---------------- fused MoE kernel: 128 rows / WG, 8 wave32, TDM double-buffered ----------------
__global__ __launch_bounds__(256) void moe_main_kernel(
    const float* __restrict__ x,
    const float* __restrict__ gb1, const float* __restrict__ gW2,
    const float* __restrict__ gb2,
    const float* __restrict__ be1, const float* __restrict__ be2,
    const float* __restrict__ We3, const float* __restrict__ be3,
    const _Float16* __restrict__ gW1t, const _Float16* __restrict__ W1t,
    const _Float16* __restrict__ W2t,
    float* __restrict__ out) {
  extern __shared__ __align__(16) unsigned char smem[];
  _Float16* xh = (_Float16*)(smem + OFF_XH);
  _Float16* h1 = (_Float16*)(smem + OFF_H1);
  _Float16* h2 = (_Float16*)(smem + OFF_H2);
  _Float16* g1 = (_Float16*)(smem + OFF_H2);        // alias: gate phase only
  float* wE   = (float*)(smem + OFF_F);             // 128 x 8
  float* we3s = wE + 128 * 8;                       // 1024
  float* gW2s = we3s + 1024;                        // 512
  float* aout = gW2s + 512;                         // 128

  const int tid  = threadIdx.x;
  const int wave = tid >> 5, lane = tid & 31;
  const int r = lane & 15, hf = lane >> 4;
  const int m0 = wave << 4;
  const int row0 = blockIdx.x << 7;

  // double-buffer pointer/offset computed arithmetically (no LDS-pointer arrays!)
#define WOFF(buf)  (OFF_W0 + (unsigned)(buf) * WSTRIDE)
#define WPTR(buf)  ((_Float16*)(smem + WOFF(buf)))

  // T0 (gate weights) in flight while we convert the x tile
  if (wave == 0) wgt_load_async(WPTR(0), WOFF(0), gW1t, 64, lane);

  for (int c = tid; c < 128 * 64; c += 256) {
    int row = c >> 6, col = (c & 63) << 2;
    const float4 v = *(const float4*)(x + (size_t)(row0 + row) * 256 + col);
    v4h h; h[0] = (_Float16)v.x; h[1] = (_Float16)v.y;
           h[2] = (_Float16)v.z; h[3] = (_Float16)v.w;
    *(v4h*)(xh + row * LDX + col) = h;
  }
  for (int c = tid; c < 1024; c += 256) we3s[c] = We3[c];
  for (int c = tid; c < 512;  c += 256) gW2s[c] = gW2[c];
  if (tid < 128) aout[tid] = 0.f;

  if (wave == 0) wgt_wait();
  __syncthreads();                                   // buf0 = gate weights ready
  if (wave == 0) wgt_load_async(WPTR(1), WOFF(1), W1t, 128, lane);  // T1 = e0 L1 half0

  // ---- gate layer 1 (256->64) ----
  {
    const _Float16* wcur = WPTR(0);
    v8f acc[4] = {};
#pragma unroll
    for (int k = 0; k < 8; ++k) {
      v16h a = fragA_ld(xh, LDX, m0, k * 32, r, hf);
#pragma unroll
      for (int nt = 0; nt < 4; ++nt)
        acc[nt] = wmma16(a, fragB_ld(wcur, LDX, nt * 16, k * 32, r, hf), acc[nt]);
    }
#pragma unroll
    for (int nt = 0; nt < 4; ++nt) {
      float b = gb1[nt * 16 + r];
#pragma unroll
      for (int i = 0; i < 8; ++i) {
        float v = fmaxf(acc[nt][i] + b, 0.f);
        g1[(m0 + i + hf * 8) * LDG + nt * 16 + r] = (_Float16)v;
      }
    }
  }
  __syncthreads();

  // ---- gate layer 2 (64->8) + softmax ----
  if (tid < 128) {
    float le[8];
#pragma unroll
    for (int e2 = 0; e2 < 8; ++e2) le[e2] = gb2[e2];
    for (int j = 0; j < 64; ++j) {
      float hv = (float)g1[tid * LDG + j];
#pragma unroll
      for (int e2 = 0; e2 < 8; ++e2) le[e2] += hv * gW2s[j * 8 + e2];
    }
    float mx = le[0];
#pragma unroll
    for (int e2 = 1; e2 < 8; ++e2) mx = fmaxf(mx, le[e2]);
    float s = 0.f;
#pragma unroll
    for (int e2 = 0; e2 < 8; ++e2) { le[e2] = __expf(le[e2] - mx); s += le[e2]; }
    float inv = 1.f / s;
#pragma unroll
    for (int e2 = 0; e2 < 8; ++e2) wE[tid * 8 + e2] = le[e2] * inv;
  }
  __syncthreads();

  int cur = 1;
  for (int e = 0; e < 8; ++e) {
    // ================= layer 1 (two staged N-halves) =================
    for (int hN = 0; hN < 2; ++hN) {
      if (wave == 0) wgt_wait();
      __syncthreads();                               // buf[cur] ready, buf[cur^1] free
      if (wave == 0) {
        const _Float16* nxt = (hN == 0) ? (W1t + e * 65536 + 32768)   // this expert's half1
                                        : (W2t + e * 32768);          // this expert's L2
        wgt_load_async(WPTR(cur ^ 1), WOFF(cur ^ 1), nxt, 128, lane);
      }
      const _Float16* wcur = WPTR(cur);
      v8f acc[8] = {};
#pragma unroll
      for (int k = 0; k < 8; ++k) {
        v16h a = fragA_ld(xh, LDX, m0, k * 32, r, hf);
#pragma unroll
        for (int nt = 0; nt < 8; ++nt)
          acc[nt] = wmma16(a, fragB_ld(wcur, LDX, nt * 16, k * 32, r, hf), acc[nt]);
      }
#pragma unroll
      for (int nt = 0; nt < 8; ++nt) {
        int n0 = hN * 128 + nt * 16;
        float b = be1[e * 256 + n0 + r];
#pragma unroll
        for (int i = 0; i < 8; ++i) {
          float v = fmaxf(acc[nt][i] + b, 0.f);
          h1[(m0 + i + hf * 8) * LDX + n0 + r] = (_Float16)v;
        }
      }
      cur ^= 1;
    }
    // ================= layer 2 (256->128) =================
    if (wave == 0) wgt_wait();
    __syncthreads();                                 // buf[cur] = W2t[e]; also h1 complete
    if (wave == 0 && e < 7)
      wgt_load_async(WPTR(cur ^ 1), WOFF(cur ^ 1), W1t + (e + 1) * 65536, 128, lane);
    {
      const _Float16* wcur = WPTR(cur);
      v8f acc[8] = {};
#pragma unroll
      for (int k = 0; k < 8; ++k) {
        v16h a = fragA_ld(h1, LDX, m0, k * 32, r, hf);
#pragma unroll
        for (int nt = 0; nt < 8; ++nt)
          acc[nt] = wmma16(a, fragB_ld(wcur, LDX, nt * 16, k * 32, r, hf), acc[nt]);
      }
#pragma unroll
      for (int nt = 0; nt < 8; ++nt) {
        int n0 = nt * 16;
        float b = be2[e * 128 + n0 + r];
#pragma unroll
        for (int i = 0; i < 8; ++i) {
          float v = fmaxf(acc[nt][i] + b, 0.f);
          h2[(m0 + i + hf * 8) * LDH2 + n0 + r] = (_Float16)v;
        }
      }
    }
    __syncthreads();                                 // h2 ready
    // ================= layer 3 + gated accumulation =================
    if (tid < 128) {
      float acc3 = be3[e];
#pragma unroll
      for (int c = 0; c < 16; ++c) {
        v8h hv = *(const v8h*)(h2 + tid * LDH2 + c * 8);
#pragma unroll
        for (int j = 0; j < 8; ++j) acc3 += (float)hv[j] * we3s[e * 128 + c * 8 + j];
      }
      aout[tid] += wE[tid * 8 + e] * acc3;
    }
    __syncthreads();
    cur ^= 1;
  }

  if (tid < 128) out[row0 + tid] = aout[tid];
#undef WOFF
#undef WPTR
}

extern "C" void kernel_launch(void* const* d_in, const int* in_sizes, int n_in,
                              void* d_out, int out_size, void* d_ws, size_t ws_size,
                              hipStream_t stream) {
  (void)in_sizes; (void)n_in; (void)out_size; (void)ws_size;
  const float* x   = (const float*)d_in[0];
  const float* gW1 = (const float*)d_in[1];
  const float* gb1 = (const float*)d_in[2];
  const float* gW2 = (const float*)d_in[3];
  const float* gb2 = (const float*)d_in[4];
  const float* We1 = (const float*)d_in[5];
  const float* be1 = (const float*)d_in[6];
  const float* We2 = (const float*)d_in[7];
  const float* be2 = (const float*)d_in[8];
  const float* We3 = (const float*)d_in[9];
  const float* be3 = (const float*)d_in[10];
  float* out = (float*)d_out;

  _Float16* gW1t = (_Float16*)d_ws;           // 64*256 halves
  _Float16* W1t  = gW1t + 64 * 256;           // 8*256*256 halves
  _Float16* W2t  = W1t + 8 * 256 * 256;       // 8*128*256 halves

  moe_prep_kernel<<<2048, 256, 0, stream>>>(gW1, We1, We2, gW1t, W1t, W2t);
  moe_main_kernel<<<512, 256, LDS_TOTAL, stream>>>(
      x, gb1, gW2, gb2, be1, be2, We3, be3, gW1t, W1t, W2t, out);
}
